// VoxelBackBone8xFuse_72636486910064
// MI455X (gfx1250) — matrix operand order repaired and checked by reading, not compiled, not run on hardware
//
#include <hip/hip_runtime.h>
#include <hip/hip_bf16.h>
#include <math.h>

// ---------------------------------------------------------------------------
// VoxelBackBone8xFuse for MI455X (gfx1250): implicit-GEMM 3D convs on WMMA
// f16 16x16x32 -> f32. Activations f16 (~140MB working set, L2-resident in
// 192MB L2). Input grid padded to Cin=8. A-fragments = 2 x 16B aligned b128
// per lane per K-chunk, loaded BRANCHLESSLY (OOB runs redirected to a zeroed
// guard page via cndmask). Cin>=32 layers use a fully-unrolled row-structured
// kernel (compile-time (kd,kh,kw) nests: no div/mod decode, hoisted bounds
// checks). Cin=8/16 layers use the generic ping-pong pipelined kernel.
// BN+ReLU+mask fused into the WMMA epilogue.
// ---------------------------------------------------------------------------

typedef __attribute__((ext_vector_type(16))) _Float16 v16h;
typedef __attribute__((ext_vector_type(8)))  _Float16 v8h;
typedef __attribute__((ext_vector_type(8)))  float    v8f;

#define IMG_H 376
#define IMG_W 1240

// ---------------- small utility kernels ----------------

__global__ void zero_u32(uint32_t* p, long n) {
  long i = (long)blockIdx.x * blockDim.x + threadIdx.x;
  long stride = (long)gridDim.x * blockDim.x;
  for (; i < n; i += stride) p[i] = 0u;
}

// scatter into Cin=8-padded dense grid (channels 4..7 stay zero)
__global__ void scatter_kernel(const float* __restrict__ feat,
                               const int* __restrict__ coords,
                               _Float16* __restrict__ dense,
                               float* __restrict__ m1, int nvox) {
  int i = blockIdx.x * blockDim.x + threadIdx.x;
  if (i >= nvox) return;
  int z = coords[i * 4 + 1], y = coords[i * 4 + 2], x = coords[i * 4 + 3];
  long s = ((long)z * 200 + y) * 176 + x;
  for (int c = 0; c < 4; ++c) dense[s * 8 + c] = (_Float16)feat[i * 4 + c];
  m1[s] = 1.0f;
}

__global__ void bn_prep(const float* __restrict__ gamma,
                        const float* __restrict__ beta,
                        const float* __restrict__ mean,
                        const float* __restrict__ var,
                        float* __restrict__ scale, float* __restrict__ bias,
                        int n) {
  int i = blockIdx.x * blockDim.x + threadIdx.x;
  if (i >= n) return;
  float s = gamma[i] * rsqrtf(var[i] + 1e-3f);
  scale[i] = s;
  bias[i]  = beta[i] - mean[i] * s;
}

// max-pool for occupancy masks (matches lax.reduce_window max)
__global__ void mask_pool(const float* __restrict__ in, float* __restrict__ out,
                          int Din, int Hin, int Win,
                          int Dout, int Hout, int Wout,
                          int KD, int KH, int KW,
                          int SD, int SH, int SW,
                          int PD, int PH, int PW) {
  int i = blockIdx.x * blockDim.x + threadIdx.x;
  int nS = Dout * Hout * Wout;
  if (i >= nS) return;
  int od = i / (Hout * Wout);
  int r  = i % (Hout * Wout);
  int oh = r / Wout, ow = r % Wout;
  float m = 0.0f;
  for (int kd = 0; kd < KD; ++kd) {
    int id = od * SD - PD + kd;
    if ((unsigned)id >= (unsigned)Din) continue;
    for (int kh = 0; kh < KH; ++kh) {
      int ih = oh * SH - PH + kh;
      if ((unsigned)ih >= (unsigned)Hin) continue;
      for (int kw = 0; kw < KW; ++kw) {
        int iw = ow * SW - PW + kw;
        if ((unsigned)iw >= (unsigned)Win) continue;
        float v = in[((long)id * Hin + ih) * Win + iw];
        m = m > v ? m : v;
      }
    }
  }
  out[i] = m;
}

// ---------------- WMMA B-layout weight pack ----------------
// B (32x16 f16) ISA layout: column n = lane&15; element e holds
// K = (e&7) + (lane>=16 ? 8 : 0) + (e>=8 ? 16 : 0). Packed as [chunk][coutTile][lane][e].
// CinPad may exceed CinReal (zero-padded input channels -> zero weights).
template <int KD, int KH, int KW>
__global__ void pack_weights(const float* __restrict__ w, _Float16* __restrict__ wp,
                             int CinPad, int CinReal, int Cout) {
  const int K = KD * KH * KW * CinPad;
  const int nChunks = (K + 31) / 32;
  const int coutTiles = Cout / 16;
  const int total = coutTiles * nChunks * 32 * 16;
  int i = blockIdx.x * blockDim.x + threadIdx.x;
  if (i >= total) return;
  int e = i & 15;
  int lane = (i >> 4) & 31;
  int ct = (i >> 9) % coutTiles;
  int c = (i >> 9) / coutTiles;
  int hi = lane >> 4;
  int n = ct * 16 + (lane & 15);
  int k = c * 32 + (e & 7) + (hi ? 8 : 0) + ((e >> 3) ? 16 : 0);
  _Float16 val = (_Float16)0.0f;
  if (k < K) {
    int cin = k % CinPad;
    int rest = k / CinPad;
    int kw = rest % KW; rest /= KW;
    int kh = rest % KH;
    int kd = rest / KH;
    if (cin < CinReal)
      val = (_Float16)w[((((long)kd * KH + kh) * KW + kw) * CinReal + cin) * (long)Cout + n];
  }
  wp[i] = val;
}

// ---------------- shared epilogue: BN + ReLU + mask + store ----------------
template <int NT>
__device__ inline void conv_epilogue(const v8f* acc,
                                     const float* __restrict__ scale,
                                     const float* __restrict__ bias,
                                     const float* __restrict__ mask,
                                     void* __restrict__ outv, int f32out,
                                     int base, int nSpat, int lane, int hi) {
  const int Cout = NT * 16;
  float sc[NT], bi[NT];
#pragma unroll
  for (int t = 0; t < NT; ++t) {
    int n = t * 16 + (lane & 15);
    sc[t] = scale[n];
    bi[t] = bias[n];
  }
#pragma unroll
  for (int r = 0; r < 8; ++r) {
    int so = base + r + (hi ? 8 : 0);
    if (so < nSpat) {
      float mk = mask[so];
#pragma unroll
      for (int t = 0; t < NT; ++t) {
        int n = t * 16 + (lane & 15);
        float v = acc[t][r] * sc[t] + bi[t];
        v = v > 0.0f ? v : 0.0f;
        v *= mk;
        if (f32out) ((float*)outv)[(long)so * Cout + n] = v;
        else        ((_Float16*)outv)[(long)so * Cout + n] = (_Float16)v;
      }
    }
  }
}

// ---------------- generic conv3d on WMMA (Cin = 8 or 16) ----------------
// Ping-pong software-pipelined K loop, branchless guarded A loads.
template <int KD, int KH, int KW, int NT>
__global__ void conv3d_wmma_gen(const _Float16* __restrict__ in,
                                const _Float16* __restrict__ wpack,
                                const float* __restrict__ scale,
                                const float* __restrict__ bias,
                                const float* __restrict__ mask,
                                void* __restrict__ outv, int f32out,
                                const _Float16* __restrict__ zg,
                                int Din, int Hin, int Win, int CinShift,
                                int Dout, int Hout, int Wout,
                                int SD, int SH, int SW,
                                int PD, int PH, int PW) {
  const int lane = threadIdx.x & 31;
  const int wave = threadIdx.x >> 5;
  const int nSpat = Dout * Hout * Wout;
  const int base = (blockIdx.x * 8 + wave) * 16;   // wave-uniform
  if (base >= nSpat) return;                       // uniform exit per wave
  const int Cin = 1 << CinShift;
  const int K = KD * KH * KW * Cin;
  const int nChunks = (K + 31) / 32;
  const int hi = lane >> 4;
  const int mrow = lane & 15;

  int s = base + mrow;
  bool rowOK = s < nSpat;
  int od = s / (Hout * Wout);
  int r0_ = s % (Hout * Wout);
  int oh = r0_ / Wout;
  int ow = r0_ % Wout;
  const int id0 = od * SD - PD, ih0 = oh * SH - PH, iw0 = ow * SW - PW;

  auto loadA = [&](int c) -> v16h {
    union { v16h v; v8h h[2]; } u;
    const int kbase = c * 32 + (hi ? 8 : 0);
#pragma unroll
    for (int rr = 0; rr < 2; ++rr) {
      const int k0 = kbase + rr * 16;
      int cin = k0 & (Cin - 1);
      int rest = k0 >> CinShift;
      int kw = rest % KW; rest /= KW;
      int kh = rest % KH;
      int kd = rest / KH;
      int id = id0 + kd, ih = ih0 + kh, iw = iw0 + kw;
      bool ok = rowOK && (k0 < K) &&
                (unsigned)id < (unsigned)Din && (unsigned)ih < (unsigned)Hin &&
                (unsigned)iw < (unsigned)Win;
      const _Float16* p =
          ok ? (in + ((((long)id * Hin + ih) * Win + iw) << CinShift) + cin) : zg;
      u.h[rr] = *(const v8h*)p;   // unconditional; OOB reads zero guard page
    }
    return u.v;
  };
  auto loadB = [&](int c, int t) -> v16h {
    return *(const v16h*)(wpack + ((long)(c * NT + t) * 32 + lane) * 16);
  };

  v8f acc[NT] = {};

  // ping-pong software pipeline: prefetch chunk c+1/c+2 while WMMAing c/c+1
  v16h a0 = loadA(0);
  v16h b0[NT];
#pragma unroll
  for (int t = 0; t < NT; ++t) b0[t] = loadB(0, t);
  v16h a1;
  v16h b1[NT];
  int c = 0;
  for (; c + 2 < nChunks; c += 2) {
    a1 = loadA(c + 1);
#pragma unroll
    for (int t = 0; t < NT; ++t) b1[t] = loadB(c + 1, t);
#pragma unroll
    for (int t = 0; t < NT; ++t)
      acc[t] = __builtin_amdgcn_wmma_f32_16x16x32_f16(
          false, a0, false, b0[t], (short)0, acc[t], false, false);
    a0 = loadA(c + 2);
#pragma unroll
    for (int t = 0; t < NT; ++t) b0[t] = loadB(c + 2, t);
#pragma unroll
    for (int t = 0; t < NT; ++t)
      acc[t] = __builtin_amdgcn_wmma_f32_16x16x32_f16(
          false, a1, false, b1[t], (short)0, acc[t], false, false);
  }
  if (c + 1 < nChunks) {            // two chunks remain
    a1 = loadA(c + 1);
#pragma unroll
    for (int t = 0; t < NT; ++t) b1[t] = loadB(c + 1, t);
#pragma unroll
    for (int t = 0; t < NT; ++t)
      acc[t] = __builtin_amdgcn_wmma_f32_16x16x32_f16(
          false, a0, false, b0[t], (short)0, acc[t], false, false);
#pragma unroll
    for (int t = 0; t < NT; ++t)
      acc[t] = __builtin_amdgcn_wmma_f32_16x16x32_f16(
          false, a1, false, b1[t], (short)0, acc[t], false, false);
  } else {                          // one chunk remains
#pragma unroll
    for (int t = 0; t < NT; ++t)
      acc[t] = __builtin_amdgcn_wmma_f32_16x16x32_f16(
          false, a0, false, b0[t], (short)0, acc[t], false, false);
  }

  conv_epilogue<NT>(acc, scale, bias, mask, outv, f32out, base, nSpat, lane, hi);
}

// ---------------- row-structured conv3d on WMMA (Cin >= 32) ----------------
// Fully unrolled compile-time (kd,kh,kw-chunk) nests: no div/mod decode,
// id/ih checks hoisted per row, iw check per kw, one pointer select per chunk
// (guard page covers both 16B runs: p0 and p0+16).
template <int KD, int KH, int KW, int NT, int CS>
__global__ void conv3d_wmma_row(const _Float16* __restrict__ in,
                                const _Float16* __restrict__ wpack,
                                const float* __restrict__ scale,
                                const float* __restrict__ bias,
                                const float* __restrict__ mask,
                                void* __restrict__ outv, int f32out,
                                const _Float16* __restrict__ zg,
                                int Din, int Hin, int Win,
                                int Dout, int Hout, int Wout,
                                int SD, int SH, int SW,
                                int PD, int PH, int PW) {
  constexpr int Cin = 1 << CS;
  constexpr int CPW = Cin / 32;          // chunks per kw (>=1)
  constexpr int CPR = KW * CPW;          // chunks per (kd,kh) row
  static_assert((KW * Cin) % 32 == 0, "row must be whole chunks");

  const int lane = threadIdx.x & 31;
  const int wave = threadIdx.x >> 5;
  const int nSpat = Dout * Hout * Wout;
  const int base = (blockIdx.x * 8 + wave) * 16;   // wave-uniform
  if (base >= nSpat) return;                       // uniform exit per wave
  const int hi = lane >> 4;
  const int mrow = lane & 15;

  int s = base + mrow;
  bool rowOK = s < nSpat;
  int od = s / (Hout * Wout);
  int r0_ = s % (Hout * Wout);
  int oh = r0_ / Wout;
  int ow = r0_ % Wout;
  const int id0 = od * SD - PD, ih0 = oh * SH - PH, iw0 = ow * SW - PW;
  const int cinHi = hi * 8;

  v8f acc[NT] = {};
  int chunk = 0;
#pragma unroll
  for (int kd = 0; kd < KD; ++kd) {
    const int id = id0 + kd;
    const bool okd = rowOK && (unsigned)id < (unsigned)Din;
#pragma unroll
    for (int kh = 0; kh < KH; ++kh) {
      const int ih = ih0 + kh;
      const bool okr = okd && (unsigned)ih < (unsigned)Hin;
      const long rowbase = ((long)id * Hin + ih) * (long)Win;
#pragma unroll
      for (int j = 0; j < CPR; ++j) {
        const int kw = j / CPW;
        const int cinb = (j % CPW) * 32 + cinHi;
        const int iw = iw0 + kw;
        const bool ok = okr && (unsigned)iw < (unsigned)Win;
        const _Float16* p0 =
            ok ? (in + ((rowbase + iw) << CS) + cinb) : zg;
        union { v16h v; v8h h[2]; } u;
        u.h[0] = *(const v8h*)(p0);        // K run 0 (8 halves)
        u.h[1] = *(const v8h*)(p0 + 16);   // K run 1 (guard page is 64B)
        const v16h a = u.v;
#pragma unroll
        for (int t = 0; t < NT; ++t) {
          v16h b = *(const v16h*)(wpack + ((long)(chunk * NT + t) * 32 + lane) * 16);
          acc[t] = __builtin_amdgcn_wmma_f32_16x16x32_f16(
              false, a, false, b, (short)0, acc[t], false, false);
        }
        ++chunk;
      }
    }
  }

  conv_epilogue<NT>(acc, scale, bias, mask, outv, f32out, base, nSpat, lane, hi);
}

// ---------------- image fusion (projective + bilinear + sigmoid gate) ------
__global__ void fuse_kernel(const int* __restrict__ coords,
                            const float* __restrict__ seg,
                            const float* __restrict__ T,   // 4x4 row-major
                            const float* __restrict__ P,   // 3x4 row-major
                            const float* __restrict__ fw,
                            const float* __restrict__ fb,
                            _Float16* __restrict__ x1, int nvox) {
  int i = blockIdx.x * blockDim.x + threadIdx.x;
  if (i >= nvox) return;
  int z = coords[i * 4 + 1], y = coords[i * 4 + 2], x = coords[i * 4 + 3];
  float cx = ((float)x + 0.5f) * 0.2f + 0.0f;
  float cy = ((float)y + 0.5f) * 0.2f - 20.0f;
  float cz = ((float)z + 0.5f) * 0.1f - 3.0f;
  float camx = T[0] * cx + T[1] * cy + T[2]  * cz + T[3];
  float camy = T[4] * cx + T[5] * cy + T[6]  * cz + T[7];
  float camz = T[8] * cx + T[9] * cy + T[10] * cz + T[11];
  float wh   = T[12] * cx + T[13] * cy + T[14] * cz + T[15];
  float iw = 1.0f / wh;
  camx *= iw; camy *= iw; camz *= iw;
  float prx = P[0] * camx + P[1] * camy + P[2]  * camz + P[3];
  float pry = P[4] * camx + P[5] * camy + P[6]  * camz + P[7];
  float prz = P[8] * camx + P[9] * camy + P[10] * camz + P[11];
  float u = prx / prz, v = pry / prz;
  float gx = u * (2.0f / IMG_W) - 1.0f;
  float gy = v * (2.0f / IMG_H) - 1.0f;
  float fx = (gx + 1.0f) * (IMG_W * 0.5f) - 0.5f;
  float fy = (gy + 1.0f) * (IMG_H * 0.5f) - 0.5f;
  float x0f = floorf(fx), y0f = floorf(fy);
  float wx = fx - x0f, wy = fy - y0f;
  int ix0 = (int)x0f, iy0 = (int)y0f;
  auto tap = [&](int yi, int xi) -> float {
    bool ok = (yi >= 0) && (yi < IMG_H) && (xi >= 0) && (xi < IMG_W);
    int yc = yi < 0 ? 0 : (yi > IMG_H - 1 ? IMG_H - 1 : yi);
    int xc = xi < 0 ? 0 : (xi > IMG_W - 1 ? IMG_W - 1 : xi);
    float l0 = seg[yc * IMG_W + xc];
    float l1 = seg[IMG_H * IMG_W + yc * IMG_W + xc];
    float p = 1.0f / (1.0f + expf(l0 - l1));   // softmax ch-1 of 2 logits
    return ok ? p : 0.0f;
  };
  float wimg = tap(iy0, ix0)     * (1.0f - wy) * (1.0f - wx)
             + tap(iy0, ix0 + 1) * (1.0f - wy) * wx
             + tap(iy0 + 1, ix0) * wy * (1.0f - wx)
             + tap(iy0 + 1, ix0 + 1) * wy * wx;
  long off = (((long)z * 200 + y) * 176 + x) * 16;
#pragma unroll
  for (int j = 0; j < 16; ++j) {
    float lw = 1.0f / (1.0f + expf(-(wimg * fw[j] + fb[j])));
    float f1 = (float)x1[off + j];
    x1[off + j] = (_Float16)(f1 * (1.0f + lw));
  }
}

// ---------------------------------------------------------------------------
// host side
// ---------------------------------------------------------------------------

namespace {

struct LayerDesc {
  int widx;                   // d_in index of 'w' (gamma/beta/mean/var follow)
  int cinPad, cinReal, cout;
  int KD, KH, KW;
  int SD, SH, SW;
  int PD, PH, PW;
};

inline size_t align256(size_t x) { return (x + 255) & ~(size_t)255; }

} // namespace

extern "C" void kernel_launch(void* const* d_in, const int* in_sizes, int n_in,
                              void* d_out, int out_size, void* d_ws, size_t ws_size,
                              hipStream_t stream) {
  (void)in_sizes; (void)n_in; (void)out_size; (void)ws_size;

  // Input indices per setup_inputs() insertion order; params dict flattened
  // in insertion order, each conv block = {w, gamma, beta, mean, var}.
  const int IN_VF = 0, IN_SEG = 1, IN_T = 2, IN_P = 3;
  const int PBASE = 4;
  const int IN_FW = PBASE + 12 * 5;      // 64
  const int IN_FB = IN_FW + 1;           // 65
  const int IN_COORD = IN_FB + 1;        // 66

  LayerDesc L[12] = {
    {PBASE + 0 * 5,  8,  4,  16, 3, 3, 3, 1, 1, 1, 1, 1, 1}, // conv_input (padded Cin)
    {PBASE + 1 * 5, 16, 16,  16, 3, 3, 3, 1, 1, 1, 1, 1, 1}, // conv1
    {PBASE + 2 * 5, 16, 16,  32, 3, 3, 3, 2, 2, 2, 1, 1, 1}, // conv2_0
    {PBASE + 3 * 5, 32, 32,  32, 3, 3, 3, 1, 1, 1, 1, 1, 1},
    {PBASE + 4 * 5, 32, 32,  32, 3, 3, 3, 1, 1, 1, 1, 1, 1},
    {PBASE + 5 * 5, 32, 32,  64, 3, 3, 3, 2, 2, 2, 1, 1, 1}, // conv3_0
    {PBASE + 6 * 5, 64, 64,  64, 3, 3, 3, 1, 1, 1, 1, 1, 1},
    {PBASE + 7 * 5, 64, 64,  64, 3, 3, 3, 1, 1, 1, 1, 1, 1},
    {PBASE + 8 * 5, 64, 64,  64, 3, 3, 3, 2, 2, 2, 0, 1, 1}, // conv4_0 pad (0,1,1)
    {PBASE + 9 * 5, 64, 64,  64, 3, 3, 3, 1, 1, 1, 1, 1, 1},
    {PBASE +10 * 5, 64, 64,  64, 3, 3, 3, 1, 1, 1, 1, 1, 1},
    {PBASE +11 * 5, 64, 64, 128, 3, 1, 1, 2, 1, 1, 0, 0, 0}  // conv_out (3,1,1)
  };

  // spatial levels
  const int D1 = 41, H1 = 200, W1 = 176;   long NS1 = (long)D1 * H1 * W1;
  const int D2 = 21, H2 = 100, W2 = 88;    long NS2 = (long)D2 * H2 * W2;
  const int D3 = 11, H3 = 50,  W3 = 44;    long NS3 = (long)D3 * H3 * W3;
  const int D4 = 5,  H4 = 25,  W4 = 22;    long NS4 = (long)D4 * H4 * W4;
  const int DO = 2,  HO = 25,  WO = 22;    long NSO = (long)DO * HO * WO;

  // workspace carve-up
  char* ws = (char*)d_ws;
  size_t off = 0;
  auto alloc = [&](size_t bytes) -> char* {
    char* p = ws + off;
    off += align256(bytes);
    return p;
  };
  _Float16* zguard = (_Float16*)alloc(64);                          // 64B zero page
  _Float16* dense0 = (_Float16*)alloc(NS1 * 8 * sizeof(_Float16));  // Cin padded to 8
  _Float16* xin    = (_Float16*)alloc(NS1 * 16 * sizeof(_Float16));
  _Float16* x1     = (_Float16*)alloc(NS1 * 16 * sizeof(_Float16));
  _Float16* x2a    = (_Float16*)alloc(NS2 * 32 * sizeof(_Float16));
  _Float16* x2b    = (_Float16*)alloc(NS2 * 32 * sizeof(_Float16));
  _Float16* x3a    = (_Float16*)alloc(NS3 * 64 * sizeof(_Float16));
  _Float16* x3b    = (_Float16*)alloc(NS3 * 64 * sizeof(_Float16));
  _Float16* x4a    = (_Float16*)alloc(NS4 * 64 * sizeof(_Float16));
  _Float16* x4b    = (_Float16*)alloc(NS4 * 64 * sizeof(_Float16));
  float* m1 = (float*)alloc(NS1 * sizeof(float));
  float* m2 = (float*)alloc(NS2 * sizeof(float));
  float* m3 = (float*)alloc(NS3 * sizeof(float));
  float* m4 = (float*)alloc(NS4 * sizeof(float));
  float* mo = (float*)alloc(NSO * sizeof(float));
  float* bnScale[12]; float* bnBias[12];
  _Float16* wpack[12];
  for (int l = 0; l < 12; ++l) {
    bnScale[l] = (float*)alloc(128 * sizeof(float));
    bnBias[l]  = (float*)alloc(128 * sizeof(float));
    int K = L[l].KD * L[l].KH * L[l].KW * L[l].cinPad;
    int nChunks = (K + 31) / 32;
    wpack[l] = (_Float16*)alloc((size_t)(L[l].cout / 16) * nChunks * 512 * sizeof(_Float16));
  }

  // 1) zero guard page + dense voxel grid + occupancy mask, scatter voxels
  {
    zero_u32<<<dim3(1), dim3(64), 0, stream>>>((uint32_t*)zguard, 16);
    long nWords = NS1 * 8 / 2;
    zero_u32<<<dim3(2048), dim3(256), 0, stream>>>((uint32_t*)dense0, nWords);
    zero_u32<<<dim3(2048), dim3(256), 0, stream>>>((uint32_t*)m1, NS1);
    int nvox = in_sizes[IN_COORD] / 4;
    scatter_kernel<<<dim3((nvox + 255) / 256), dim3(256), 0, stream>>>(
        (const float*)d_in[IN_VF], (const int*)d_in[IN_COORD], dense0, m1, nvox);
  }

  // 2) fold BN params, pack weights into WMMA B layout
  for (int l = 0; l < 12; ++l) {
    const float* w     = (const float*)d_in[L[l].widx + 0];
    const float* gamma = (const float*)d_in[L[l].widx + 1];
    const float* beta  = (const float*)d_in[L[l].widx + 2];
    const float* mean  = (const float*)d_in[L[l].widx + 3];
    const float* var   = (const float*)d_in[L[l].widx + 4];
    bn_prep<<<dim3(1), dim3(128), 0, stream>>>(gamma, beta, mean, var,
                                               bnScale[l], bnBias[l], L[l].cout);
    int K = L[l].KD * L[l].KH * L[l].KW * L[l].cinPad;
    int total = (L[l].cout / 16) * ((K + 31) / 32) * 512;
    if (L[l].KH == 3)
      pack_weights<3, 3, 3><<<dim3((total + 255) / 256), dim3(256), 0, stream>>>(
          w, wpack[l], L[l].cinPad, L[l].cinReal, L[l].cout);
    else
      pack_weights<3, 1, 1><<<dim3((total + 255) / 256), dim3(256), 0, stream>>>(
          w, wpack[l], L[l].cinPad, L[l].cinReal, L[l].cout);
  }

  // 3) occupancy mask pyramid
  mask_pool<<<dim3(((int)NS2 + 255) / 256), dim3(256), 0, stream>>>(
      m1, m2, D1, H1, W1, D2, H2, W2, 3, 3, 3, 2, 2, 2, 1, 1, 1);
  mask_pool<<<dim3(((int)NS3 + 255) / 256), dim3(256), 0, stream>>>(
      m2, m3, D2, H2, W2, D3, H3, W3, 3, 3, 3, 2, 2, 2, 1, 1, 1);
  mask_pool<<<dim3(((int)NS4 + 255) / 256), dim3(256), 0, stream>>>(
      m3, m4, D3, H3, W3, D4, H4, W4, 3, 3, 3, 2, 2, 2, 0, 1, 1);
  mask_pool<<<dim3(((int)NSO + 255) / 256), dim3(256), 0, stream>>>(
      m4, mo, D4, H4, W4, DO, HO, WO, 3, 1, 1, 2, 1, 1, 0, 0, 0);

  // 4) conv chain on WMMA (grid.x over spatial tiles; cout register-blocked)
  auto gridFor = [](long nSpat) {
    int nTiles = (int)((nSpat + 15) / 16);
    return dim3((nTiles + 7) / 8);
  };
  auto cs = [](int cin) { return 31 - __builtin_clz((unsigned)cin); };

#define LGEN(KDv, KHv, KWv, NTv, l, src, dst, f32o, msk, Di, Hi, Wi, Do, Ho, Wo) \
  conv3d_wmma_gen<KDv, KHv, KWv, NTv><<<gridFor((long)(Do) * (Ho) * (Wo)), dim3(256), 0, stream>>>( \
      src, wpack[l], bnScale[l], bnBias[l], msk, dst, f32o, zguard, \
      Di, Hi, Wi, cs(L[l].cinPad), Do, Ho, Wo, \
      L[l].SD, L[l].SH, L[l].SW, L[l].PD, L[l].PH, L[l].PW)
#define LROW(KDv, KHv, KWv, NTv, CSv, l, src, dst, f32o, msk, Di, Hi, Wi, Do, Ho, Wo) \
  conv3d_wmma_row<KDv, KHv, KWv, NTv, CSv><<<gridFor((long)(Do) * (Ho) * (Wo)), dim3(256), 0, stream>>>( \
      src, wpack[l], bnScale[l], bnBias[l], msk, dst, f32o, zguard, \
      Di, Hi, Wi, Do, Ho, Wo, \
      L[l].SD, L[l].SH, L[l].SW, L[l].PD, L[l].PH, L[l].PW)

  LGEN(3, 3, 3, 1, 0, dense0, xin, 0, m1, D1, H1, W1, D1, H1, W1);   // conv_input Cin=8
  LGEN(3, 3, 3, 1, 1, xin,    x1,  0, m1, D1, H1, W1, D1, H1, W1);   // conv1 Cin=16

  // image fusion applied in-place on x1 at voxel positions
  {
    int nvox = in_sizes[IN_COORD] / 4;
    fuse_kernel<<<dim3((nvox + 255) / 256), dim3(256), 0, stream>>>(
        (const int*)d_in[IN_COORD], (const float*)d_in[IN_SEG],
        (const float*)d_in[IN_T], (const float*)d_in[IN_P],
        (const float*)d_in[IN_FW], (const float*)d_in[IN_FB], x1, nvox);
  }

  LGEN(3, 3, 3, 2, 2, x1,  x2a, 0, m2, D1, H1, W1, D2, H2, W2);      // conv2_0 Cin=16 s2
  LROW(3, 3, 3, 2, 5, 3, x2a, x2b, 0, m2, D2, H2, W2, D2, H2, W2);   // conv2_1 Cin=32
  LROW(3, 3, 3, 2, 5, 4, x2b, x2a, 0, m2, D2, H2, W2, D2, H2, W2);   // conv2_2 Cin=32
  LROW(3, 3, 3, 4, 5, 5, x2a, x3a, 0, m3, D2, H2, W2, D3, H3, W3);   // conv3_0 Cin=32 s2
  LROW(3, 3, 3, 4, 6, 6, x3a, x3b, 0, m3, D3, H3, W3, D3, H3, W3);   // conv3_1 Cin=64
  LROW(3, 3, 3, 4, 6, 7, x3b, x3a, 0, m3, D3, H3, W3, D3, H3, W3);   // conv3_2
  LROW(3, 3, 3, 4, 6, 8, x3a, x4a, 0, m4, D3, H3, W3, D4, H4, W4);   // conv4_0 s2
  LROW(3, 3, 3, 4, 6, 9, x4a, x4b, 0, m4, D4, H4, W4, D4, H4, W4);   // conv4_1
  LROW(3, 3, 3, 4, 6, 10, x4b, x4a, 0, m4, D4, H4, W4, D4, H4, W4);  // conv4_2
  LROW(3, 1, 1, 8, 6, 11, x4a, d_out, 1, mo, D4, H4, W4, DO, HO, WO); // conv_out Cin=64 -> f32
#undef LGEN
#undef LROW
}
